// MVRBind_36026185679171
// MI455X (gfx1250) — matrix operand order repaired and checked by reading, not compiled, not run on hardware
//
#include <hip/hip_runtime.h>
#include <cstdint>
#include <cstddef>

// ---------------------------------------------------------------------------
// CDNA5 (gfx1250): bf16 WMMA GEMMs (32x64 tile/wave, 8 wmma per K-step),
// coalesced GCN scatter, fused elementwise tail.
// ---------------------------------------------------------------------------

typedef __attribute__((ext_vector_type(8)))  __bf16 v8bf;
typedef __attribute__((ext_vector_type(16))) __bf16 v16bf;
typedef __attribute__((ext_vector_type(8)))  float  v8f;

#define NODES 50000
#define DIM   256
#define DIM3  768
#define WPB   8   // waves per block in the GEMM kernel

// -------------------------- input index map --------------------------------
enum InIdx {
  I_X = 0,
  I_CONV1_W, I_CONV1_B, I_CONV2_W, I_CONV2_B, I_CONV3_W, I_CONV3_B,
  I_C1SE_W, I_C1SE_B, I_C2SE_W, I_C2SE_B, I_C3SE_W, I_C3SE_B,
  I_C1F_W,  I_C1F_B,  I_C2F_W,  I_C2F_B,  I_C3F_W,  I_C3F_B,
  I_LC1_W,  I_LC1_B,  I_LC2_W,  I_LC2_B,  I_LC3_W,  I_LC3_B,
  I_Q_W, I_Q_B, I_K_W, I_K_B, I_V_W, I_V_B,
  I_AO_W, I_AO_B, I_L4_W, I_L4_B, I_L1_W, I_L1_B, I_L2_W, I_L2_B,
  I_TV_W, I_TV_B, I_TO_W, I_TO_B, I_TF1_W, I_TF1_B, I_TF2_W, I_TF2_B,
  I_TLN1_G, I_TLN1_B, I_TLN2_G, I_TLN2_B,
  I_BN_G, I_BN_B, I_BN_RM, I_BN_RV,
  I_EI, I_EIS, I_EIF
};

// -------------------------- utility kernels --------------------------------

__device__ __forceinline__ unsigned hash_u32(unsigned v) {
  v = v * 747796405u + 2891336453u;
  unsigned w = ((v >> ((v >> 28u) + 4u)) ^ v) * 277803737u;
  return (w >> 22u) ^ w;
}

__global__ void k_noise(const float* __restrict__ x, float* __restrict__ o, long n) {
  long i = (long)blockIdx.x * blockDim.x + threadIdx.x;
  if (i >= n) return;
  unsigned r1 = hash_u32((unsigned)(2 * i + 1));
  unsigned r2 = hash_u32((unsigned)(2 * i + 2));
  float u1 = ((float)r1 + 1.0f) * 2.3283064e-10f;
  float u2 = (float)r2 * 2.3283064e-10f;
  float g = sqrtf(-2.0f * logf(u1)) * cosf(6.28318530718f * u2);
  o[i] = x[i] + 0.05f * g;
}

__global__ void k_fill1(float* __restrict__ p, int n) {
  int i = blockIdx.x * blockDim.x + threadIdx.x;
  if (i < n) p[i] = 1.0f;
}

__global__ void k_deg(const int* __restrict__ dst, float* __restrict__ deg, int E) {
  int e = blockIdx.x * blockDim.x + threadIdx.x;
  if (e >= E) return;
  atomicAdd(&deg[dst[e]], 1.0f);
}

__global__ void k_edge_norm(const int* __restrict__ s, const int* __restrict__ d,
                            const float* __restrict__ deg, float* __restrict__ nrm, int E) {
  int e = blockIdx.x * blockDim.x + threadIdx.x;
  if (e >= E) return;
  nrm[e] = rsqrtf(deg[s[e]]) * rsqrtf(deg[d[e]]);
}

__global__ void k_fill_bias(float* __restrict__ o, const float* __restrict__ b,
                            long n, int D) {
  long i = (long)blockIdx.x * blockDim.x + threadIdx.x;
  if (i >= n) return;
  o[i] = b[i % D];
}

// one block per edge, one thread per feature: coalesced gather + f32 atomics
__global__ void k_scatter(const float* __restrict__ h, const int* __restrict__ s,
                          const int* __restrict__ d, const float* __restrict__ nrm,
                          float* __restrict__ out, int E, int D) {
  int e = blockIdx.x;
  if (e >= E) return;
  int si = s[e], di = d[e];
  float w = nrm[e];
  int f = threadIdx.x;
  atomicAdd(&out[(size_t)di * D + f], h[(size_t)si * D + f] * w);
}

__global__ void k_selfloop_relu(float* __restrict__ out, const float* __restrict__ h,
                                const float* __restrict__ deg, long n, int D, int relu) {
  long i = (long)blockIdx.x * blockDim.x + threadIdx.x;
  if (i >= n) return;
  long row = i / D;
  float v = out[i] + h[i] / deg[row];   // self-loop norm = 1/deg
  if (relu) v = fmaxf(v, 0.0f);
  out[i] = v;
}

__global__ void k_to_bf16(const float* __restrict__ in, __bf16* __restrict__ out, long n) {
  long i = (long)blockIdx.x * blockDim.x + threadIdx.x;
  if (i < n) out[i] = (__bf16)in[i];
}

// transpose-convert W[K,Nc] f32 -> Wt[Nc,K] bf16 (B fragments become contiguous)
__global__ void k_wt(const float* __restrict__ W, __bf16* __restrict__ Wt, int K, int Nc) {
  int i = blockIdx.x * blockDim.x + threadIdx.x;
  if (i >= K * Nc) return;
  int k = i / Nc, c = i % Nc;
  Wt[(size_t)c * K + k] = (__bf16)W[i];
}

// -------------------------- WMMA GEMM --------------------------------------
// C[M,Nc] = A[M,K](bf16,row-major) @ Bt[Nc,K](bf16) + bias, optional ReLU.
// One wave computes a 32x64 tile: 2 A-fragments x 4 B-fragments -> 8
// v_wmma_f32_16x16x32_bf16 per K-step. Row guard is done by CLAMPING the A
// row (OOB A-rows only feed OOB D-rows, which are never stored) so the inner
// loop has no EXEC masking, just wide loads + wmma.
// A fragment (16x32): lane L (row=L&15) loads K {b..b+7},{b+8..b+15},
// b=(L>=16)*16 -> two contiguous 16B loads.
// B fragment (32x16): lane L (col=L&15) loads K=(L>=16)*16+0..15 -> one 32B load.
#define WMMA_BF16(Acc, Af, Bf) \
  Acc = __builtin_amdgcn_wmma_f32_16x16x32_bf16(false, Af, false, Bf, (short)0, Acc, false, false)

__global__ __launch_bounds__(32 * WPB)
void k_gemm_wmma_bf16(const __bf16* __restrict__ A, const __bf16* __restrict__ Bt,
                      const float* __restrict__ bias, float* __restrict__ C,
                      int M, int K, int Nc, int relu) {
  const int lane = threadIdx.x & 31;
  const int wave = threadIdx.y;
  const int mt = (M + 31) >> 5;     // 32-row tiles
  const int nt = Nc >> 6;           // 64-col tiles
  int tile = blockIdx.x * WPB + wave;
  if (tile >= mt * nt) return;
  const int tm = tile % mt;
  const int tn = tile / mt;
  const int half = lane >> 4;
  const int l16  = lane & 15;

  int r0 = tm * 32 + l16;       if (r0 >= M) r0 = M - 1;   // clamp (store-guarded)
  int r1 = tm * 32 + 16 + l16;  if (r1 >= M) r1 = M - 1;
  const int c0 = tn * 64 + l16;

  const __bf16* Ap0 = A + (size_t)r0 * K + half * 16;
  const __bf16* Ap1 = A + (size_t)r1 * K + half * 16;
  const __bf16* Bp0 = Bt + (size_t)(c0)      * K + half * 16;
  const __bf16* Bp1 = Bt + (size_t)(c0 + 16) * K + half * 16;
  const __bf16* Bp2 = Bt + (size_t)(c0 + 32) * K + half * 16;
  const __bf16* Bp3 = Bt + (size_t)(c0 + 48) * K + half * 16;

  v8f a00 = {}, a01 = {}, a02 = {}, a03 = {};
  v8f a10 = {}, a11 = {}, a12 = {}, a13 = {};

  for (int kk = 0; kk < K; kk += 32) {
    v16bf af0, af1;
    ((v8bf*)&af0)[0] = *(const v8bf*)(Ap0 + kk);
    ((v8bf*)&af0)[1] = *(const v8bf*)(Ap0 + kk + 8);
    ((v8bf*)&af1)[0] = *(const v8bf*)(Ap1 + kk);
    ((v8bf*)&af1)[1] = *(const v8bf*)(Ap1 + kk + 8);
    v16bf b0 = *(const v16bf*)(Bp0 + kk);
    v16bf b1 = *(const v16bf*)(Bp1 + kk);
    v16bf b2 = *(const v16bf*)(Bp2 + kk);
    v16bf b3 = *(const v16bf*)(Bp3 + kk);
    if (kk + 32 < K) {
      __builtin_prefetch(Ap0 + kk + 32, 0, 1);   // global_prefetch_b8
      __builtin_prefetch(Bp0 + kk + 32, 0, 1);
    }
    WMMA_BF16(a00, af0, b0);
    WMMA_BF16(a01, af0, b1);
    WMMA_BF16(a02, af0, b2);
    WMMA_BF16(a03, af0, b3);
    WMMA_BF16(a10, af1, b0);
    WMMA_BF16(a11, af1, b1);
    WMMA_BF16(a12, af1, b2);
    WMMA_BF16(a13, af1, b3);
  }

  // store: C layout per tile: VGPR v -> row (v + half*8), col = l16
  const int rb0 = tm * 32 + half * 8;
  const int rb1 = rb0 + 16;
  float bv0 = 0.f, bv1 = 0.f, bv2 = 0.f, bv3 = 0.f;
  if (bias) {
    bv0 = bias[c0]; bv1 = bias[c0 + 16]; bv2 = bias[c0 + 32]; bv3 = bias[c0 + 48];
  }
#pragma unroll
  for (int v = 0; v < 8; ++v) {
    int row0 = rb0 + v, row1 = rb1 + v;
    if (row0 < M) {
      float* Cr = C + (size_t)row0 * Nc + c0;
      float x0 = a00[v] + bv0, x1 = a01[v] + bv1, x2 = a02[v] + bv2, x3 = a03[v] + bv3;
      if (relu) { x0 = fmaxf(x0, 0.f); x1 = fmaxf(x1, 0.f); x2 = fmaxf(x2, 0.f); x3 = fmaxf(x3, 0.f); }
      Cr[0] = x0; Cr[16] = x1; Cr[32] = x2; Cr[48] = x3;
    }
    if (row1 < M) {
      float* Cr = C + (size_t)row1 * Nc + c0;
      float x0 = a10[v] + bv0, x1 = a11[v] + bv1, x2 = a12[v] + bv2, x3 = a13[v] + bv3;
      if (relu) { x0 = fmaxf(x0, 0.f); x1 = fmaxf(x1, 0.f); x2 = fmaxf(x2, 0.f); x3 = fmaxf(x3, 0.f); }
      Cr[0] = x0; Cr[16] = x1; Cr[32] = x2; Cr[48] = x3;
    }
  }
}

// -------------------------- layernorm --------------------------------------
__global__ void k_add_ln(const float* __restrict__ A, const float* __restrict__ B,
                         const float* __restrict__ G, const float* __restrict__ Be,
                         float* __restrict__ O, int D3v) {
  long row = blockIdx.x;
  const float* a = A + row * (long)D3v;
  const float* b = B + row * (long)D3v;
  float s = 0.f, s2 = 0.f;
  for (int f = threadIdx.x; f < D3v; f += blockDim.x) {
    float v = a[f] + b[f];
    s += v; s2 += v * v;
  }
  __shared__ float sh[64];
  for (int o = 16; o > 0; o >>= 1) {
    s  += __shfl_down(s, o, 32);
    s2 += __shfl_down(s2, o, 32);
  }
  int wid = threadIdx.x >> 5;
  if ((threadIdx.x & 31) == 0) { sh[wid] = s; sh[wid + 32] = s2; }
  __syncthreads();
  if (threadIdx.x == 0) {
    float ts = 0.f, ts2 = 0.f;
    int nw = blockDim.x >> 5;
    for (int i = 0; i < nw; ++i) { ts += sh[i]; ts2 += sh[i + 32]; }
    sh[0] = ts; sh[1] = ts2;
  }
  __syncthreads();
  float mean = sh[0] / D3v;
  float var  = sh[1] / D3v - mean * mean;
  float inv  = rsqrtf(var + 1e-5f);
  float* o = O + row * (long)D3v;
  for (int f = threadIdx.x; f < D3v; f += blockDim.x) {
    float v = a[f] + b[f];
    o[f] = (v - mean) * inv * G[f] + Be[f];
  }
}

// -------------------------- shape shufflers --------------------------------

__global__ void k_concat3(const float* __restrict__ a, const float* __restrict__ b,
                          const float* __restrict__ c, float* __restrict__ o, long n) {
  long i = (long)blockIdx.x * blockDim.x + threadIdx.x;
  if (i >= n) return;
  long row = i / DIM3;
  int f = (int)(i % DIM3);
  float v;
  if (f < 256)      v = a[row * 256 + f];
  else if (f < 512) v = b[row * 256 + (f - 256)];
  else              v = c[row * 256 + (f - 512)];
  o[i] = v;
}

__global__ void k_stack4(const float* __restrict__ a, const float* __restrict__ b,
                         const float* __restrict__ c, const float* __restrict__ d,
                         float* __restrict__ o, long n) {
  long i = (long)blockIdx.x * blockDim.x + threadIdx.x;
  if (i >= n) return;
  long node = i >> 10;
  int r = (int)(i & 1023);
  int q = r >> 8;
  int f = r & 255;
  const float* s = (q == 0) ? a : (q == 1) ? b : (q == 2) ? c : d;
  o[i] = s[node * 256 + f];
}

// per-node 4-token attention
__global__ void k_attn_pool(const float* __restrict__ Q, const float* __restrict__ K,
                            const float* __restrict__ V, float* __restrict__ O, int N) {
  int n = blockIdx.x * blockDim.x + threadIdx.x;
  if (n >= N) return;
  const float* q = Q + (size_t)n * 512;
  const float* k = K + (size_t)n * 512;
  const float* v = V + (size_t)n * 512;
  float a[4][4];
  for (int qi = 0; qi < 4; ++qi)
    for (int ki = 0; ki < 4; ++ki) {
      float s = 0.f;
      for (int dd = 0; dd < 128; ++dd) s += q[qi * 128 + dd] * k[ki * 128 + dd];
      a[qi][ki] = s * 0.08838834764831845f;   // 1/sqrt(128)
    }
  for (int qi = 0; qi < 4; ++qi) {
    float m = fmaxf(fmaxf(a[qi][0], a[qi][1]), fmaxf(a[qi][2], a[qi][3]));
    float sum = 0.f;
    for (int ki = 0; ki < 4; ++ki) { a[qi][ki] = expf(a[qi][ki] - m); sum += a[qi][ki]; }
    float r = 1.0f / sum;
    for (int ki = 0; ki < 4; ++ki) a[qi][ki] *= r;
  }
  float* o = O + (size_t)n * 512;
  for (int qi = 0; qi < 4; ++qi)
    for (int dd = 0; dd < 128; ++dd) {
      float s = a[qi][0] * v[dd] + a[qi][1] * v[128 + dd] +
                a[qi][2] * v[256 + dd] + a[qi][3] * v[384 + dd];
      o[qi * 128 + dd] = s;
    }
}

__global__ void k_bn_relu(const float* __restrict__ h, const float* __restrict__ g,
                          const float* __restrict__ b, const float* __restrict__ rm,
                          const float* __restrict__ rv, float* __restrict__ o, long n) {
  long i = (long)blockIdx.x * blockDim.x + threadIdx.x;
  if (i >= n) return;
  int f = (int)(i % 128);
  float v = (h[i] - rm[f]) * rsqrtf(rv[f] + 1e-5f) * g[f] + b[f];
  o[i] = fmaxf(v, 0.0f);
}

__global__ void k_final_sigmoid(const float* __restrict__ h, const float* __restrict__ w2,
                                const float* __restrict__ b2, float* __restrict__ out, int N) {
  int n = blockIdx.x * blockDim.x + threadIdx.x;
  if (n >= N) return;
  float s = b2[0];
  const float* hr = h + (size_t)n * 128;
  for (int f = 0; f < 128; ++f) s += hr[f] * w2[f];
  out[n] = 1.0f / (1.0f + expf(-s));
}

// -------------------------- host-side helpers ------------------------------

static inline long cdiv(long a, long b) { return (a + b - 1) / b; }

static void gemm_core(hipStream_t st, const __bf16* Abf, const __bf16* Wt,
                      const float* bias, float* C, int M, int K, int Nc, int relu) {
  int mt = (M + 31) / 32, nt = Nc / 64;
  int blocks = (mt * nt + WPB - 1) / WPB;
  k_gemm_wmma_bf16<<<dim3((unsigned)blocks), dim3(32, WPB), 0, st>>>(
      Abf, Wt, bias, C, M, K, Nc, relu);
}

static void run_gemm(hipStream_t st, __bf16* bfA, const float* Af32, const __bf16* Wt,
                     const float* bias, float* C, int M, int K, int Nc, int relu) {
  long na = (long)M * K;
  k_to_bf16<<<dim3((unsigned)cdiv(na, 256)), dim3(256), 0, st>>>(Af32, bfA, na);
  gemm_core(st, bfA, Wt, bias, C, M, K, Nc, relu);
}

static void run_gcn(hipStream_t st, __bf16* bfA, const float* xin, int fin,
                    const __bf16* Wt, const float* b,
                    const int* src, const int* dst, const float* nrm, const float* deg,
                    int E, float* hbuf, float* out, int relu) {
  run_gemm(st, bfA, xin, Wt, nullptr, hbuf, NODES, fin, DIM, 0);   // h = x @ W
  long n = (long)NODES * DIM;
  k_fill_bias<<<dim3((unsigned)cdiv(n, 256)), dim3(256), 0, st>>>(out, b, n, DIM);
  k_scatter<<<dim3((unsigned)E), dim3(DIM), 0, st>>>(hbuf, src, dst, nrm, out, E, DIM);
  k_selfloop_relu<<<dim3((unsigned)cdiv(n, 256)), dim3(256), 0, st>>>(out, hbuf, deg, n, DIM, relu);
}

static void prep_graph(hipStream_t st, const int* src, const int* dst, int E,
                       float* deg, float* nrm) {
  k_fill1<<<dim3((unsigned)cdiv(NODES, 256)), dim3(256), 0, st>>>(deg, NODES);
  k_deg<<<dim3((unsigned)cdiv(E, 256)), dim3(256), 0, st>>>(dst, deg, E);
  k_edge_norm<<<dim3((unsigned)cdiv(E, 256)), dim3(256), 0, st>>>(src, dst, deg, nrm, E);
}

// -------------------------- entry point ------------------------------------

extern "C" void kernel_launch(void* const* d_in, const int* in_sizes, int n_in,
                              void* d_out, int out_size, void* d_ws, size_t ws_size,
                              hipStream_t stream) {
  (void)n_in; (void)out_size; (void)ws_size;
  const float* x = (const float*)d_in[I_X];
#define FP(idx) ((const float*)d_in[idx])
  const int* ei  = (const int*)d_in[I_EI];
  const int* eis = (const int*)d_in[I_EIS];
  const int* eif = (const int*)d_in[I_EIF];
  const int E  = in_sizes[I_EI]  / 2;
  const int Es = in_sizes[I_EIS] / 2;
  const int Ef = in_sizes[I_EIF] / 2;

  // ---- workspace bump allocator ----
  char* wp = (char*)d_ws;
  auto alloc_f = [&](size_t elems) -> float* {
    float* p = (float*)wp;
    wp += (elems * sizeof(float) + 255) & ~(size_t)255;
    return p;
  };
  auto alloc_h = [&](size_t elems) -> __bf16* {
    __bf16* p = (__bf16*)wp;
    wp += (elems * sizeof(__bf16) + 255) & ~(size_t)255;
    return p;
  };
  const size_t ND  = (size_t)NODES * DIM;
  const size_t ND3 = (size_t)NODES * DIM3;

  float* xb   = alloc_f(ND);
  float* hbuf = alloc_f(ND);
  float* s1 = alloc_f(ND); float* s2 = alloc_f(ND); float* s3 = alloc_f(ND);
  float* f1 = alloc_f(ND); float* f2 = alloc_f(ND); float* f3 = alloc_f(ND);
  float* m1 = alloc_f(ND); float* m2 = alloc_f(ND); float* m3 = alloc_f(ND);
  float* C1 = alloc_f(ND3); float* C2 = alloc_f(ND3); float* C3 = alloc_f(ND3);
  float* T1 = alloc_f(ND3); float* T2 = alloc_f(ND3);
  float* tok = alloc_f((size_t)NODES * 1024);
  float* Qb = alloc_f((size_t)NODES * 512);
  float* Kb = alloc_f((size_t)NODES * 512);
  float* Vb = alloc_f((size_t)NODES * 512);
  float* Ob = alloc_f((size_t)NODES * 512);
  float* x5 = alloc_f(ND);
  float* h4 = alloc_f(ND);
  float* h1 = alloc_f((size_t)NODES * 128);
  float* h2 = alloc_f((size_t)NODES * 128);
  float* degE = alloc_f(NODES); float* degS = alloc_f(NODES); float* degF = alloc_f(NODES);
  float* nrmE = alloc_f(E);     float* nrmS = alloc_f(Es);    float* nrmF = alloc_f(Ef);
  __bf16* bfA = alloc_h((size_t)4 * NODES * 256);   // activation staging

  // ---- pre-convert + transpose ALL weights to bf16 once ----
  auto make_wt = [&](int idx, int K, int Nc) -> __bf16* {
    __bf16* p = alloc_h((size_t)K * Nc);
    int nw = K * Nc;
    k_wt<<<dim3((unsigned)cdiv(nw, 256)), dim3(256), 0, stream>>>(FP(idx), p, K, Nc);
    return p;
  };
  __bf16* W_c1   = make_wt(I_CONV1_W, 256, 256);
  __bf16* W_c2   = make_wt(I_CONV2_W, 768, 256);
  __bf16* W_c3   = make_wt(I_CONV3_W, 768, 256);
  __bf16* W_c1s  = make_wt(I_C1SE_W, 256, 256);
  __bf16* W_c2s  = make_wt(I_C2SE_W, 256, 256);
  __bf16* W_c3s  = make_wt(I_C3SE_W, 256, 256);
  __bf16* W_c1f  = make_wt(I_C1F_W, 256, 256);
  __bf16* W_c2f  = make_wt(I_C2F_W, 256, 256);
  __bf16* W_c3f  = make_wt(I_C3F_W, 256, 256);
  __bf16* W_lc1  = make_wt(I_LC1_W, 768, 256);
  __bf16* W_lc2  = make_wt(I_LC2_W, 768, 256);
  __bf16* W_lc3  = make_wt(I_LC3_W, 768, 256);
  __bf16* W_q    = make_wt(I_Q_W, 256, 128);
  __bf16* W_k    = make_wt(I_K_W, 256, 128);
  __bf16* W_v    = make_wt(I_V_W, 256, 128);
  __bf16* W_ao   = make_wt(I_AO_W, 512, 256);
  __bf16* W_l4   = make_wt(I_L4_W, 256, 256);
  __bf16* W_l1   = make_wt(I_L1_W, 256, 128);
  __bf16* W_tv   = make_wt(I_TV_W, 768, 768);
  __bf16* W_to   = make_wt(I_TO_W, 768, 768);
  __bf16* W_tf1  = make_wt(I_TF1_W, 768, 128);
  __bf16* W_tf2  = make_wt(I_TF2_W, 128, 768);

  // x = x + 0.05 * noise
  k_noise<<<dim3((unsigned)cdiv((long)ND, 256)), dim3(256), 0, stream>>>(x, xb, (long)ND);

  // GCN symmetric norms
  prep_graph(stream, ei,  ei  + E,  E,  degE, nrmE);
  prep_graph(stream, eis, eis + Es, Es, degS, nrmS);
  prep_graph(stream, eif, eif + Ef, Ef, degF, nrmF);

  // structural / sequence branch
  run_gcn(stream, bfA, xb, DIM, W_c1s, FP(I_C1SE_B), eis, eis + Es, nrmS, degS, Es, hbuf, s1, 1);
  run_gcn(stream, bfA, s1, DIM, W_c2s, FP(I_C2SE_B), eis, eis + Es, nrmS, degS, Es, hbuf, s2, 1);
  run_gcn(stream, bfA, s2, DIM, W_c3s, FP(I_C3SE_B), eis, eis + Es, nrmS, degS, Es, hbuf, s3, 1);
  // window-graph branch
  run_gcn(stream, bfA, xb, DIM, W_c1f, FP(I_C1F_B), eif, eif + Ef, nrmF, degF, Ef, hbuf, f1, 1);
  run_gcn(stream, bfA, f1, DIM, W_c2f, FP(I_C2F_B), eif, eif + Ef, nrmF, degF, Ef, hbuf, f2, 1);
  run_gcn(stream, bfA, f2, DIM, W_c3f, FP(I_C3F_B), eif, eif + Ef, nrmF, degF, Ef, hbuf, f3, 1);

  // transformer helper (weights shared across the 3 applications)
  auto transformer = [&](float* Cio) {
    run_gemm(stream, bfA, Cio, W_tv, FP(I_TV_B), T1, NODES, DIM3, DIM3, 0);
    run_gemm(stream, bfA, T1,  W_to, FP(I_TO_B), T2, NODES, DIM3, DIM3, 0);
    k_add_ln<<<dim3(NODES), dim3(256), 0, stream>>>(Cio, T2, FP(I_TLN1_G), FP(I_TLN1_B), T1, DIM3);
    run_gemm(stream, bfA, T1, W_tf1, FP(I_TF1_B), T2, NODES, DIM3, 128, 1);
    run_gemm(stream, bfA, T2, W_tf2, FP(I_TF2_B), Cio, NODES, 128, DIM3, 0);
    k_add_ln<<<dim3(NODES), dim3(256), 0, stream>>>(T1, Cio, FP(I_TLN2_G), FP(I_TLN2_B), Cio, DIM3);
  };

  const long n768 = (long)ND3;
  run_gcn(stream, bfA, xb, DIM, W_c1, FP(I_CONV1_B), ei, ei + E, nrmE, degE, E, hbuf, m1, 1);
  k_concat3<<<dim3((unsigned)cdiv(n768, 256)), dim3(256), 0, stream>>>(m1, s1, f1, C1, n768);
  transformer(C1);
  run_gcn(stream, bfA, C1, DIM3, W_c2, FP(I_CONV2_B), ei, ei + E, nrmE, degE, E, hbuf, m2, 1);
  k_concat3<<<dim3((unsigned)cdiv(n768, 256)), dim3(256), 0, stream>>>(m2, s2, f2, C2, n768);
  transformer(C2);
  run_gcn(stream, bfA, C2, DIM3, W_c3, FP(I_CONV3_B), ei, ei + E, nrmE, degE, E, hbuf, m3, 1);
  k_concat3<<<dim3((unsigned)cdiv(n768, 256)), dim3(256), 0, stream>>>(m3, s3, f3, C3, n768);
  transformer(C3);

  // level projections (overwrite m1..m3 with x1..x3)
  run_gemm(stream, bfA, C1, W_lc1, FP(I_LC1_B), m1, NODES, DIM3, DIM, 1);
  run_gemm(stream, bfA, C2, W_lc2, FP(I_LC2_B), m2, NODES, DIM3, DIM, 1);
  run_gemm(stream, bfA, C3, W_lc3, FP(I_LC3_B), m3, NODES, DIM3, DIM, 1);

  // four-level attention pooling: convert token batch once, 3 GEMMs off it
  long ntok = (long)NODES * 1024;
  k_stack4<<<dim3((unsigned)cdiv(ntok, 256)), dim3(256), 0, stream>>>(xb, m1, m2, m3, tok, ntok);
  k_to_bf16<<<dim3((unsigned)cdiv(ntok, 256)), dim3(256), 0, stream>>>(tok, bfA, ntok);
  gemm_core(stream, bfA, W_q, FP(I_Q_B), Qb, 4 * NODES, 256, 128, 0);
  gemm_core(stream, bfA, W_k, FP(I_K_B), Kb, 4 * NODES, 256, 128, 0);
  gemm_core(stream, bfA, W_v, FP(I_V_B), Vb, 4 * NODES, 256, 128, 0);
  k_attn_pool<<<dim3((unsigned)cdiv(NODES, 128)), dim3(128), 0, stream>>>(Qb, Kb, Vb, Ob, NODES);

  // head MLP
  run_gemm(stream, bfA, Ob, W_ao, FP(I_AO_B), x5, NODES, 512, 256, 0);
  run_gemm(stream, bfA, x5, W_l4, FP(I_L4_B), h4, NODES, 256, 256, 1);
  run_gemm(stream, bfA, h4, W_l1, FP(I_L1_B), h1, NODES, 256, 128, 0);
  long nh = (long)NODES * 128;
  k_bn_relu<<<dim3((unsigned)cdiv(nh, 256)), dim3(256), 0, stream>>>(
      h1, FP(I_BN_G), FP(I_BN_B), FP(I_BN_RM), FP(I_BN_RV), h2, nh);
  k_final_sigmoid<<<dim3((unsigned)cdiv(NODES, 256)), dim3(256), 0, stream>>>(
      h2, FP(I_L2_W), FP(I_L2_B), (float*)d_out, NODES);
#undef FP
}
